// Conditioner_43482248904953
// MI455X (gfx1250) — matrix operand order repaired and checked by reading, not compiled, not run on hardware
//
#include <hip/hip_runtime.h>
#include <hip/hip_bf16.h>

typedef __attribute__((ext_vector_type(16))) _Float16 v16h;
typedef __attribute__((ext_vector_type(8)))  _Float16 v8h;
typedef __attribute__((ext_vector_type(8)))  float    v8f;

// Problem dimensions (fixed by the reference)
constexpr int Bsz    = 8;
constexpr int N_OBJ  = 8192;
constexpr int Mpts   = 8192;
constexpr int N_TOK  = 1024;
constexpr int D_LAT  = 1024;
constexpr int D_FEAT = 512;
constexpr int D_OUT  = 1024;
constexpr int Ktot   = D_LAT + D_FEAT; // 1536

// ---------------------------------------------------------------------------
// Kernel 1: nearest[i] = argmin_j || obj_pts0[i] - precomputed_points[j] ||^2
// ---------------------------------------------------------------------------
__global__ __launch_bounds__(256) void nearest_kernel(
    const float* __restrict__ obj_surf,   // (B, N_OBJ, 6)
    const float* __restrict__ pts,        // (M, 3)
    int* __restrict__ nearest)            // (N_OBJ)
{
    __shared__ float sp[256 * 3];
    const int i = blockIdx.x * 256 + threadIdx.x;
    const float qx = obj_surf[(size_t)i * 6 + 0];
    const float qy = obj_surf[(size_t)i * 6 + 1];
    const float qz = obj_surf[(size_t)i * 6 + 2];
    float best = 3.4e38f;
    int   bestj = 0;
    for (int j0 = 0; j0 < Mpts; j0 += 256) {
        for (int t = threadIdx.x; t < 256 * 3; t += 256)
            sp[t] = pts[(size_t)j0 * 3 + t];
        __syncthreads();
        #pragma unroll 4
        for (int jj = 0; jj < 256; ++jj) {
            const float dx = qx - sp[jj * 3 + 0];
            const float dy = qy - sp[jj * 3 + 1];
            const float dz = qz - sp[jj * 3 + 2];
            const float d = dx * dx + dy * dy + dz * dz;
            if (d < best) { best = d; bestj = j0 + jj; }
        }
        __syncthreads();
    }
    nearest[i] = bestj;
}

// ---------------------------------------------------------------------------
// Kernel 2: per (b, t) dual argmin over obj_pts[b]; fold nearest[] so the
// output is the final feature-row index used by the GEMM gather.
// ---------------------------------------------------------------------------
__global__ __launch_bounds__(256) void token_idx_kernel(
    const float* __restrict__ obj_surf,    // (B, N_OBJ, 6)
    const float* __restrict__ gpc,         // (B, N_TOK, 3) -> use [0]
    const float* __restrict__ lpc,         // (B, N_TOK, 3) -> use [0]
    const int*   __restrict__ nearest,     // (N_OBJ)
    int* __restrict__ fidx_g,              // (B * N_TOK)
    int* __restrict__ fidx_l)              // (B * N_TOK)
{
    __shared__ float sp[256 * 3];
    const int t = blockIdx.x * 256 + threadIdx.x;
    const int b = blockIdx.y;
    const float gx = gpc[(size_t)t * 3 + 0], gy = gpc[(size_t)t * 3 + 1], gz = gpc[(size_t)t * 3 + 2];
    const float lx = lpc[(size_t)t * 3 + 0], ly = lpc[(size_t)t * 3 + 1], lz = lpc[(size_t)t * 3 + 2];
    float bg = 3.4e38f, bl = 3.4e38f;
    int   ig = 0,       il = 0;
    const size_t base = (size_t)b * N_OBJ;
    for (int j0 = 0; j0 < N_OBJ; j0 += 256) {
        for (int u = threadIdx.x; u < 256 * 3; u += 256) {
            const int r = u / 3, c = u % 3;
            sp[u] = obj_surf[(base + j0 + r) * 6 + c];
        }
        __syncthreads();
        #pragma unroll 4
        for (int jj = 0; jj < 256; ++jj) {
            const float px = sp[jj * 3 + 0], py = sp[jj * 3 + 1], pz = sp[jj * 3 + 2];
            float dx = gx - px, dy = gy - py, dz = gz - pz;
            const float dg = dx * dx + dy * dy + dz * dz;
            if (dg < bg) { bg = dg; ig = j0 + jj; }
            dx = lx - px; dy = ly - py; dz = lz - pz;
            const float dl = dx * dx + dy * dy + dz * dz;
            if (dl < bl) { bl = dl; il = j0 + jj; }
        }
        __syncthreads();
    }
    fidx_g[(size_t)b * N_TOK + t] = nearest[ig];
    fidx_l[(size_t)b * N_TOK + t] = nearest[il];
}

// ---------------------------------------------------------------------------
// Kernel 3: out = [cond | feats[fidx]] @ W + bias   (f16 WMMA, f32 accum)
// 256 thr = 8 wave32, block tile 64(M) x 128(N), wave tile 32x32, K step 32.
// Double-buffered LDS: 1 barrier per K-step; tile i+1 global loads + converts
// are issued before the WMMAs of tile i. f32->f16 conversion during staging.
// ---------------------------------------------------------------------------
#define LDSPITCH 40   // halves; 80B row pitch: 16B-aligned, conflict-spread

__device__ __forceinline__ v8f wmma_f16(v16h a, v16h b, v8f c) {
    return __builtin_amdgcn_wmma_f32_16x16x32_f16(false, a, false, b,
                                                  (short)0, c, false, false);
}

__device__ __forceinline__ v8h cvt8(const float* __restrict__ p) {
    const float4 fa = ((const float4*)p)[0];
    const float4 fb = ((const float4*)p)[1];
    v8h h;
    h[0] = (_Float16)fa.x; h[1] = (_Float16)fa.y;
    h[2] = (_Float16)fa.z; h[3] = (_Float16)fa.w;
    h[4] = (_Float16)fb.x; h[5] = (_Float16)fb.y;
    h[6] = (_Float16)fb.z; h[7] = (_Float16)fb.w;
    return h;
}

// Load a 16-half fragment as two aligned 16B LDS reads (ds_load_b128).
__device__ __forceinline__ v16h ld16(const _Float16* p, int o1, int o2) {
    const v8h lo = *(const v8h*)(p + o1);
    const v8h hi = *(const v8h*)(p + o2);
    return __builtin_shufflevector(lo, hi, 0, 1, 2, 3, 4, 5, 6, 7,
                                           8, 9, 10, 11, 12, 13, 14, 15);
}

struct GemmShared {
    alignas(16) _Float16 As[2][64][LDSPITCH];    // [buf][row][k]
    alignas(16) _Float16 Bs[2][128][LDSPITCH];   // [buf][n][k] (transposed)
    int fr[64];
};

__global__ __launch_bounds__(256) void gemm_kernel(
    const float* __restrict__ cond_obj,   // (8192, 1024)
    const float* __restrict__ cond_geo,   // (8192, 1024)
    const float* __restrict__ feats,      // (8192, 512)
    const int*   __restrict__ fidx_g,     // (8192)
    const int*   __restrict__ fidx_l,     // (8192)
    const float* __restrict__ W_obj,      // (1536, 1024)
    const float* __restrict__ W_geo,      // (1536, 1024)
    const float* __restrict__ b_obj,      // (1024)
    const float* __restrict__ b_geo,      // (1024)
    float* __restrict__ out_all)          // (2, 8192, 1024)
{
    const int z = blockIdx.z;
    const float* __restrict__ cond = z ? cond_geo : cond_obj;
    const float* __restrict__ W    = z ? W_geo    : W_obj;
    const float* __restrict__ bias = z ? b_geo    : b_obj;
    const int*   __restrict__ fidx = z ? fidx_l   : fidx_g;
    float* __restrict__ out = out_all + (size_t)z * (size_t)Bsz * N_TOK * D_OUT;

    __shared__ GemmShared sh;

    const int tid  = threadIdx.x;
    const int row0 = blockIdx.y * 64;
    const int col0 = blockIdx.x * 128;

    if (tid < 64) sh.fr[tid] = fidx[row0 + tid];

    // ---- staging geometry (loop-invariant)
    const int sr  = tid >> 2;            // A: row 0..63
    const int skc = (tid & 3) << 3;      // A: k sub-offset 0,8,16,24
    const float* __restrict__ aCond = cond + (size_t)(row0 + sr) * D_LAT + skc;

    const int bn4 = (tid & 31) << 2;     // B: n base 0,4,...,124
    const int bkk = tid >> 5;            // B: k row 0..7 (per i-chunk)
    const float* __restrict__ wBase = W + col0 + bn4;

    // ---- compute geometry
    const int lane  = tid & 31;
    const int wave  = tid >> 5;
    const int wm    = wave >> 2;         // 0..1
    const int wn    = wave & 3;          // 0..3
    const int laneM = lane & 15;
    const int hi    = lane >> 4;
    const int khiA  = hi << 3;           // A: K 0..7 vs 8..15 (+16 for hi half)
    const int khiB  = hi << 3;           // B: K 0..7 vs 8..15 (+16 for hi half)
    const int aOff0 = (wm * 32 + laneM) * LDSPITCH;        // halves
    const int aOff1 = aOff0 + 16 * LDSPITCH;
    const int bOff0 = (wn * 32 + laneM) * LDSPITCH;
    const int bOff1 = bOff0 + 16 * LDSPITCH;

    __syncthreads();   // sh.fr visible to everyone
    const float* __restrict__ aFeat = feats + (size_t)sh.fr[sr] * D_FEAT + skc;

    v8f c00 = {}, c01 = {}, c10 = {}, c11 = {};

    auto stage = [&](int k0, int buf) {
        // A tile: one 16B LDS store per thread (two float4 loads + cvt)
        const float* __restrict__ ap =
            (k0 < D_LAT) ? (aCond + k0) : (aFeat + (k0 - D_LAT));
        *(v8h*)&sh.As[buf][sr][skc] = cvt8(ap);
        // B tile: 4 coalesced float4 loads of W, transposed b16 stores
        #pragma unroll
        for (int i = 0; i < 4; ++i) {
            const int kk = bkk + i * 8;
            const float4 w4 = *(const float4*)(wBase + (size_t)(k0 + kk) * D_OUT);
            sh.Bs[buf][bn4 + 0][kk] = (_Float16)w4.x;
            sh.Bs[buf][bn4 + 1][kk] = (_Float16)w4.y;
            sh.Bs[buf][bn4 + 2][kk] = (_Float16)w4.z;
            sh.Bs[buf][bn4 + 3][kk] = (_Float16)w4.w;
        }
    };

    auto compute = [&](int buf) {
        const _Float16* __restrict__ aB = &sh.As[buf][0][0];
        const _Float16* __restrict__ bB = &sh.Bs[buf][0][0];
        const v16h a0  = ld16(aB + aOff0, khiA, khiA + 16);
        const v16h a1  = ld16(aB + aOff1, khiA, khiA + 16);
        const v16h bb0 = ld16(bB + bOff0, khiB, khiB + 16);
        const v16h bb1 = ld16(bB + bOff1, khiB, khiB + 16);
        c00 = wmma_f16(a0, bb0, c00);
        c01 = wmma_f16(a0, bb1, c01);
        c10 = wmma_f16(a1, bb0, c10);
        c11 = wmma_f16(a1, bb1, c11);
    };

    // ---- software-pipelined main loop (48 K-steps, 1 barrier each)
    stage(0, 0);
    int cur = 0;
    for (int k0 = 0; k0 < Ktot; k0 += 32) {
        __syncthreads();                       // stage(cur) complete
        if (k0 + 32 < Ktot) stage(k0 + 32, cur ^ 1);
        compute(cur);
        cur ^= 1;
    }

    // ---- epilogue: D layout (N = lane&15, M = vgpr + 8*hi), add bias, NT
    const int cb0 = col0 + wn * 32 + laneM;
    const int cb1 = cb0 + 16;
    const float bv0 = bias[cb0];
    const float bv1 = bias[cb1];
    const int rb0 = row0 + wm * 32 + (hi << 3);
    const int rb1 = rb0 + 16;
    #pragma unroll
    for (int r = 0; r < 8; ++r) {
        __builtin_nontemporal_store(c00[r] + bv0, &out[(size_t)(rb0 + r) * D_OUT + cb0]);
        __builtin_nontemporal_store(c01[r] + bv1, &out[(size_t)(rb0 + r) * D_OUT + cb1]);
        __builtin_nontemporal_store(c10[r] + bv0, &out[(size_t)(rb1 + r) * D_OUT + cb0]);
        __builtin_nontemporal_store(c11[r] + bv1, &out[(size_t)(rb1 + r) * D_OUT + cb1]);
    }
}

// ---------------------------------------------------------------------------
extern "C" void kernel_launch(void* const* d_in, const int* in_sizes, int n_in,
                              void* d_out, int out_size, void* d_ws, size_t ws_size,
                              hipStream_t stream) {
    const float* object_surface   = (const float*)d_in[0];  // (8,8192,6)
    const float* precomp_points   = (const float*)d_in[1];  // (8192,3)
    const float* precomp_feats    = (const float*)d_in[2];  // (8192,512)
    const float* geo_cond         = (const float*)d_in[3];  // (8,1024,1024)
    const float* obj_cond         = (const float*)d_in[4];  // (8,1024,1024)
    const float* local_pc_infos   = (const float*)d_in[5];  // (8,1024,3)
    const float* global_pc_infos  = (const float*)d_in[6];  // (8,1024,3)
    const float* W_obj            = (const float*)d_in[7];  // (1536,1024)
    const float* b_obj            = (const float*)d_in[8];  // (1024)
    const float* W_geo            = (const float*)d_in[9];  // (1536,1024)
    const float* b_geo            = (const float*)d_in[10]; // (1024)
    float* out = (float*)d_out;

    int* nearest = (int*)d_ws;                 // 8192 ints
    int* fidx_g  = nearest + N_OBJ;            // 8192 ints
    int* fidx_l  = fidx_g + Bsz * N_TOK;       // 8192 ints

    nearest_kernel<<<dim3(N_OBJ / 256), dim3(256), 0, stream>>>(
        object_surface, precomp_points, nearest);

    token_idx_kernel<<<dim3(N_TOK / 256, Bsz), dim3(256), 0, stream>>>(
        object_surface, global_pc_infos, local_pc_infos, nearest, fidx_g, fidx_l);

    gemm_kernel<<<dim3(D_OUT / 128, (Bsz * N_TOK) / 64, 2), dim3(256), 0, stream>>>(
        obj_cond, geo_cond, precomp_feats, fidx_g, fidx_l,
        W_obj, W_geo, b_obj, b_geo, out);
}
